// CausalSelfAttention_48902497632718
// MI455X (gfx1250) — matrix operand order repaired and checked by reading, not compiled, not run on hardware
//
#include <hip/hip_runtime.h>
#include <hip/hip_bf16.h>
#include <math.h>

typedef __attribute__((ext_vector_type(16))) _Float16 v16h;
typedef __attribute__((ext_vector_type(8)))  _Float16 v8h;
typedef __attribute__((ext_vector_type(8)))  float    v8f;

#define S_LEN   2048
#define D_MODEL 1024
#define HD      64
#define NH      16
#define NTOK    4096   /* B * S */
#define QKV_N   3072

__device__ __forceinline__ v8f wmma16(v16h a, v16h b, v8f c) {
  return __builtin_amdgcn_wmma_f32_16x16x32_f16(false, a, false, b, (short)0, c,
                                                false, false);
}

// A-fragment (16x32 f16), A row-major with leading dim ld (in elements).
// Lane L: row M = L&15, K = g*16 + 0..15 contiguous, g = L>>4  -> 2x b128 loads.
__device__ __forceinline__ v16h load_A_frag(const _Float16* base, int ld, int lane) {
  const int m = lane & 15, g = lane >> 4;
  const _Float16* p = base + (size_t)m * ld + g * 16;
  v8h lo = *(const v8h*)(p);
  v8h hi = *(const v8h*)(p + 8);
  v16h a;
#pragma unroll
  for (int i = 0; i < 8; ++i) { a[i] = lo[i]; a[i + 8] = hi[i]; }
  return a;
}

// B-fragment (32x16 f16) from a K-contiguous ("transposed") matrix:
// lane L: col N = L&15 -> row of B^T, K = g*16 + e contiguous -> 32-byte load.
__device__ __forceinline__ v16h load_B_frag(const _Float16* baseT, int ld, int lane) {
  const int n = lane & 15, g = lane >> 4;
  return *(const v16h*)(baseT + (size_t)n * ld + g * 16);
}

/* -------------------- CDNA5 async global->LDS staging -------------------- */

#define WAIT_ASYNC0() asm volatile("s_wait_asynccnt 0x0" ::: "memory")
#define WAIT_DS0()    asm volatile("s_wait_dscnt 0x0" ::: "memory")

// ASYNCcnt-tracked 16B copy: LDS byte address in a VGPR, 64-bit global VGPR pair.
__device__ __forceinline__ void async_b128(unsigned int lds, const void* g) {
  asm volatile("global_load_async_to_lds_b128 %0, %1, off"
               :: "v"(lds), "v"(g) : "memory");
}

// Stage a 128-row x 32-halfword (64B/row) tile into LDS (GEMM path).
// 256 threads, 2 threads/row, 2 async b128 per thread.
__device__ __forceinline__ void stage_tile(const _Float16* __restrict__ src, int ld,
                                           int row_base, int k0,
                                           _Float16* dstLds, int tid) {
  const int crow = tid >> 1;
  const int boff = (tid & 1) * 32;  // byte offset within row
  const char* g = (const char*)(src + (size_t)(row_base + crow) * ld + k0) + boff;
  // generic shared ptr low 32 bits == LDS byte offset (flat = {aperture, lds_off})
  unsigned int l = (unsigned int)(uintptr_t)(dstLds + crow * 32) + boff;
  async_b128(l, g);
  async_b128(l + 16, g + 16);
}

// Attention: stage K tile (32 keys x 64 dims, 128B/row) and V^T tile
// (64 dims x 32 keys, 64B/row); one async b128 each per thread.
__device__ __forceinline__ void stage_kv(const _Float16* __restrict__ kbase,
                                         const _Float16* __restrict__ vbase,
                                         int key0, _Float16* Kl, _Float16* Vl,
                                         int tid) {
  {
    const int row = tid >> 3, seg = (tid & 7) * 16;
    const char* gsrc = (const char*)(kbase + (size_t)(key0 + row) * HD) + seg;
    unsigned int l = (unsigned int)(uintptr_t)(Kl + row * HD) + seg;
    async_b128(l, gsrc);
  }
  {
    const int row = tid >> 2, seg = (tid & 3) * 16;
    const char* gsrc = (const char*)(vbase + (size_t)row * S_LEN + key0) + seg;
    unsigned int l = (unsigned int)(uintptr_t)(Vl + row * 32) + seg;
    async_b128(l, gsrc);
  }
}

/* ---------------- precision conversion / transpose kernels ---------------- */

__global__ void cvt_f16_kernel(const float* __restrict__ src,
                               _Float16* __restrict__ dst, int n) {
  int i = blockIdx.x * blockDim.x + threadIdx.x;
  if (i < n) dst[i] = (_Float16)src[i];
}

// dst[cols][rows] = (f16) src[rows][cols]
__global__ void cvt_f16_T_kernel(const float* __restrict__ src,
                                 _Float16* __restrict__ dst, int rows, int cols) {
  int i = blockIdx.x * blockDim.x + threadIdx.x;
  if (i < rows * cols) {
    int r = i / cols, c = i % cols;
    dst[(size_t)c * rows + r] = (_Float16)src[i];
  }
}

/* ---------------- QKV projection + RoPE epilogue ----------------
 * Block = 256 thr = 8 waves; block tile 128x128; wave tile 32x64 (8 acc).
 * K loop double-buffers 128x32 A/B tiles in LDS via async copies.
 */
__global__ void __launch_bounds__(256)
qkv_rope_kernel(const _Float16* __restrict__ xh,
                const _Float16* __restrict__ wT,
                const float* __restrict__ rcos,
                const float* __restrict__ rsin,
                _Float16* __restrict__ qh,
                _Float16* __restrict__ kh,
                _Float16* __restrict__ vT) {
  __shared__ _Float16 As[2][128 * 32];
  __shared__ _Float16 Bs[2][128 * 32];

  const int tid = threadIdx.x;
  const int lane = tid & 31, wave = tid >> 5;
  const int wrow = (wave >> 1) * 32;    // 0,32,64,96
  const int wcol = (wave & 1) * 64;     // 0,64
  const int row0 = blockIdx.x * 128;    // token rows
  const int col0 = blockIdx.y * 128;    // qkv feature cols

  stage_tile(xh, D_MODEL, row0, 0, As[0], tid);
  stage_tile(wT, D_MODEL, col0, 0, Bs[0], tid);

  v8f acc[2][4] = {};
  for (int kt = 0; kt < 32; ++kt) {
    const int cur = kt & 1;
    WAIT_ASYNC0();
    __syncthreads();
    v16h afr[2], bfr[4];
    afr[0] = load_A_frag(&As[cur][(wrow + 0) * 32], 32, lane);
    afr[1] = load_A_frag(&As[cur][(wrow + 16) * 32], 32, lane);
#pragma unroll
    for (int t = 0; t < 4; ++t)
      bfr[t] = load_B_frag(&Bs[cur][(wcol + t * 16) * 32], 32, lane);
    WAIT_DS0();           // LDS reads drained before buffers are overwritten
    __syncthreads();
    if (kt + 1 < 32) {    // overlap next tile's DMA with the WMMAs below
      stage_tile(xh, D_MODEL, row0, (kt + 1) * 32, As[1 - cur], tid);
      stage_tile(wT, D_MODEL, col0, (kt + 1) * 32, Bs[1 - cur], tid);
    }
#pragma unroll
    for (int mi = 0; mi < 2; ++mi)
#pragma unroll
      for (int t = 0; t < 4; ++t) acc[mi][t] = wmma16(afr[mi], bfr[t], acc[mi][t]);
  }

  const int g = lane >> 4, nloc = lane & 15;
#pragma unroll
  for (int mi = 0; mi < 2; ++mi) {
#pragma unroll
    for (int t = 0; t < 4; ++t) {
      const int col   = col0 + wcol + t * 16 + nloc;  // 0..3071
      const int which = col >> 10;                    // 0=q 1=k 2=v (tile-uniform)
      const int feat  = col & (D_MODEL - 1);
      const int h = feat >> 6, d = feat & (HD - 1);
#pragma unroll
      for (int r = 0; r < 8; ++r) {
        const int row = row0 + wrow + mi * 16 + r + 8 * g;  // token in [0,4096)
        const int bb = row >> 11, s = row & (S_LEN - 1);
        float val = acc[mi][t][r];
        if (which == 2) {
          // V transposed: vT[b][h][d][s]
          vT[(((size_t)(bb * NH + h)) * HD + d) * S_LEN + s] = (_Float16)val;
        } else {
          // RoPE: pair (2t,2t+1) lives in lane^1
          float other = __shfl_xor(val, 1, 32);
          int   tt = d >> 1;
          float c  = rcos[s * (HD / 2) + tt];
          float sn = rsin[s * (HD / 2) + tt];
          float res = (d & 1) ? (other * sn + val * c)
                              : (val * c - other * sn);
          _Float16* dst = (which == 0) ? qh : kh;
          dst[(((size_t)(bb * NH + h)) * S_LEN + s) * HD + d] = (_Float16)res;
        }
      }
    }
  }
}

/* ---------------- flash-style causal attention ----------------
 * Block = 128 q rows of one (b,h); wave w owns 16 rows. K/V tiles (32 keys)
 * are cooperatively async-copied into double-buffered LDS and shared by all
 * 8 waves (8x less global traffic). Online softmax per wave; P staged via
 * per-wave LDS slab for the C->A layout transpose.
 */
__global__ void __launch_bounds__(256)
attn_kernel(const _Float16* __restrict__ qh,
            const _Float16* __restrict__ kh,
            const _Float16* __restrict__ vT,
            _Float16* __restrict__ ah) {
  __shared__ _Float16 Ks[2][32 * 64];      // 2 x 4 KB
  __shared__ _Float16 Vs[2][64 * 32];      // 2 x 4 KB
  __shared__ _Float16 Plds[8 * 16 * 32];   // 8 KB

  const int tid = threadIdx.x;
  const int lane = tid & 31, wave = tid >> 5;
  const int bh = blockIdx.x >> 4;          // b*NH + h
  const int q0 = (blockIdx.x & 15) * 128;  // block q base
  const int g = lane >> 4, nloc = lane & 15;
  const int qrow0 = q0 + wave * 16;        // this wave's q rows

  const _Float16* qbase = qh + ((size_t)bh * S_LEN + qrow0) * HD;
  const _Float16* kbase = kh + (size_t)bh * S_LEN * HD;
  const _Float16* vbase = vT + (size_t)bh * HD * S_LEN;
  _Float16* pw = Plds + wave * (16 * 32);

  const v16h qa0 = load_A_frag(qbase, HD, lane);       // k-dims 0..31
  const v16h qa1 = load_A_frag(qbase + 32, HD, lane);  // k-dims 32..63

  v8f o[4] = {};                                       // 16 x 64 output
  float mrow[8], lrow[8];
#pragma unroll
  for (int r = 0; r < 8; ++r) { mrow[r] = -__builtin_inff(); lrow[r] = 0.0f; }

  const float scale = 0.125f;              // hd^-0.5
  const int my_hi = qrow0 + 15;            // wave's causal limit
  const int nkb = ((q0 + 127) >> 5) + 1;   // block's 32-key step count

  stage_kv(kbase, vbase, 0, Ks[0], Vs[0], tid);

  for (int kb = 0; kb < nkb; ++kb) {
    const int cur = kb & 1;
    const int key0 = kb * 32;
    WAIT_ASYNC0();
    __syncthreads();
    const bool active = (key0 <= my_hi);   // wave-uniform
    v16h bk00, bk01, bk10, bk11, bv[4];
    if (active) {
      bk00 = load_B_frag(&Ks[cur][0], HD, lane);            // keys 0-15,  d 0-31
      bk01 = load_B_frag(&Ks[cur][32], HD, lane);           // keys 0-15,  d 32-63
      bk10 = load_B_frag(&Ks[cur][16 * HD], HD, lane);      // keys 16-31, d 0-31
      bk11 = load_B_frag(&Ks[cur][16 * HD + 32], HD, lane); // keys 16-31, d 32-63
#pragma unroll
      for (int t = 0; t < 4; ++t)
        bv[t] = load_B_frag(&Vs[cur][(t * 16) * 32], 32, lane);
    }
    WAIT_DS0();
    __syncthreads();
    if (kb + 1 < nkb)
      stage_kv(kbase, vbase, (kb + 1) * 32, Ks[1 - cur], Vs[1 - cur], tid);
    if (!active) continue;

    // ---- scores: q @ k^T over hd=64 ----
    v8f s0 = {}, s1 = {};
    s0 = wmma16(qa0, bk00, s0);
    s0 = wmma16(qa1, bk01, s0);
    s1 = wmma16(qa0, bk10, s1);
    s1 = wmma16(qa1, bk11, s1);

    // ---- causal mask + online softmax update ----
    float fac[8];
#pragma unroll
    for (int r = 0; r < 8; ++r) {
      const int qrow = qrow0 + r + 8 * g;
      float a0 = s0[r] * scale;
      float a1 = s1[r] * scale;
      if (key0 + nloc > qrow)      a0 = -__builtin_inff();
      if (key0 + 16 + nloc > qrow) a1 = -__builtin_inff();
      float mx = fmaxf(a0, a1);
      mx = fmaxf(mx, __shfl_xor(mx, 8, 16));
      mx = fmaxf(mx, __shfl_xor(mx, 4, 16));
      mx = fmaxf(mx, __shfl_xor(mx, 2, 16));
      mx = fmaxf(mx, __shfl_xor(mx, 1, 16));
      float mnew = fmaxf(mrow[r], mx);
      float f = __expf(mrow[r] - mnew);
      float p0 = __expf(a0 - mnew);
      float p1 = __expf(a1 - mnew);
      float rs = p0 + p1;
      rs += __shfl_xor(rs, 8, 16);
      rs += __shfl_xor(rs, 4, 16);
      rs += __shfl_xor(rs, 2, 16);
      rs += __shfl_xor(rs, 1, 16);
      lrow[r] = lrow[r] * f + rs;
      mrow[r] = mnew;
      fac[r] = f;
      const int prow = r + 8 * g;
      pw[prow * 32 + nloc]      = (_Float16)p0;
      pw[prow * 32 + 16 + nloc] = (_Float16)p1;
    }
#pragma unroll
    for (int t = 0; t < 4; ++t)
#pragma unroll
      for (int r = 0; r < 8; ++r) o[t][r] *= fac[r];

    // ---- O += P @ V ----
    v16h pa = load_A_frag(pw, 32, lane);
#pragma unroll
    for (int t = 0; t < 4; ++t) o[t] = wmma16(pa, bv[t], o[t]);
  }

  // ---- normalize and emit to [B, S, D] (f16) for the output GEMM ----
  const int b = bh >> 4, h = bh & 15;
#pragma unroll
  for (int r = 0; r < 8; ++r) {
    const float inv = 1.0f / lrow[r];
    const int s = qrow0 + r + 8 * g;
    _Float16* dst = ah + ((size_t)(b * S_LEN + s)) * D_MODEL + h * HD;
#pragma unroll
    for (int t = 0; t < 4; ++t) dst[t * 16 + nloc] = (_Float16)(o[t][r] * inv);
  }
}

/* ---------------- output projection: out = attn @ W_out (fp32 out) -------- */
__global__ void __launch_bounds__(256)
out_gemm_kernel(const _Float16* __restrict__ ah,
                const _Float16* __restrict__ wT,
                float* __restrict__ out) {
  __shared__ _Float16 As[2][128 * 32];
  __shared__ _Float16 Bs[2][128 * 32];

  const int tid = threadIdx.x;
  const int lane = tid & 31, wave = tid >> 5;
  const int wrow = (wave >> 1) * 32;
  const int wcol = (wave & 1) * 64;
  const int row0 = blockIdx.x * 128;
  const int col0 = blockIdx.y * 128;

  stage_tile(ah, D_MODEL, row0, 0, As[0], tid);
  stage_tile(wT, D_MODEL, col0, 0, Bs[0], tid);

  v8f acc[2][4] = {};
  for (int kt = 0; kt < 32; ++kt) {
    const int cur = kt & 1;
    WAIT_ASYNC0();
    __syncthreads();
    v16h afr[2], bfr[4];
    afr[0] = load_A_frag(&As[cur][(wrow + 0) * 32], 32, lane);
    afr[1] = load_A_frag(&As[cur][(wrow + 16) * 32], 32, lane);
#pragma unroll
    for (int t = 0; t < 4; ++t)
      bfr[t] = load_B_frag(&Bs[cur][(wcol + t * 16) * 32], 32, lane);
    WAIT_DS0();
    __syncthreads();
    if (kt + 1 < 32) {
      stage_tile(ah, D_MODEL, row0, (kt + 1) * 32, As[1 - cur], tid);
      stage_tile(wT, D_MODEL, col0, (kt + 1) * 32, Bs[1 - cur], tid);
    }
#pragma unroll
    for (int mi = 0; mi < 2; ++mi)
#pragma unroll
      for (int t = 0; t < 4; ++t) acc[mi][t] = wmma16(afr[mi], bfr[t], acc[mi][t]);
  }

  const int g = lane >> 4, nloc = lane & 15;
#pragma unroll
  for (int mi = 0; mi < 2; ++mi)
#pragma unroll
    for (int t = 0; t < 4; ++t)
#pragma unroll
      for (int r = 0; r < 8; ++r)
        out[(size_t)(row0 + wrow + mi * 16 + r + 8 * g) * D_MODEL +
            col0 + wcol + t * 16 + nloc] = acc[mi][t][r];
}

/* ------------------------------- launcher -------------------------------- */
extern "C" void kernel_launch(void* const* d_in, const int* in_sizes, int n_in,
                              void* d_out, int out_size, void* d_ws, size_t ws_size,
                              hipStream_t stream) {
  const float* x    = (const float*)d_in[0];   // [2,2048,1024]
  const float* rcos = (const float*)d_in[1];   // [2048,32]
  const float* rsin = (const float*)d_in[2];   // [2048,32]
  const float* wqkv = (const float*)d_in[3];   // [1024,3072]
  const float* wout = (const float*)d_in[4];   // [1024,1024]
  float* out = (float*)d_out;                  // [2,2048,1024] fp32

  char* ws = (char*)d_ws;
  _Float16* xh    = (_Float16*)(ws);                        // 8 MB
  _Float16* wqkvT = (_Float16*)(ws + (size_t)8  * 1048576); // 6 MB (transposed)
  _Float16* woutT = (_Float16*)(ws + (size_t)14 * 1048576); // 2 MB (transposed)
  _Float16* qh    = (_Float16*)(ws + (size_t)16 * 1048576); // 8 MB [B,H,S,hd]
  _Float16* kh    = (_Float16*)(ws + (size_t)24 * 1048576); // 8 MB [B,H,S,hd]
  _Float16* vT    = (_Float16*)(ws + (size_t)32 * 1048576); // 8 MB [B,H,hd,S]
  _Float16* ah    = (_Float16*)(ws + (size_t)40 * 1048576); // 8 MB [B,S,D]

  const int nx = NTOK * D_MODEL;       // 4194304
  const int nq = D_MODEL * QKV_N;      // 3145728
  const int no = D_MODEL * D_MODEL;    // 1048576

  cvt_f16_kernel<<<nx / 256, 256, 0, stream>>>(x, xh, nx);
  cvt_f16_T_kernel<<<nq / 256, 256, 0, stream>>>(wqkv, wqkvT, D_MODEL, QKV_N);
  cvt_f16_T_kernel<<<no / 256, 256, 0, stream>>>(wout, woutT, D_MODEL, D_MODEL);

  qkv_rope_kernel<<<dim3(NTOK / 128, QKV_N / 128), 256, 0, stream>>>(
      xh, wqkvT, rcos, rsin, qh, kh, vT);

  attn_kernel<<<2 * NH * (S_LEN / 128), 256, 0, stream>>>(qh, kh, vT, ah);

  out_gemm_kernel<<<dim3(NTOK / 128, D_MODEL / 128), 256, 0, stream>>>(ah, woutT, out);
}
